// AdaConv2d_64175401337378
// MI455X (gfx1250) — matrix operand order repaired and checked by reading, not compiled, not run on hardware
//
#include <hip/hip_runtime.h>

// ---------------------------------------------------------------------------
// AdaConv2d for MI455X (gfx1250, wave32, WMMA).
//
// Stage 1: per-(b,c) instance-norm stats + depthwise 3x3 (reflect) + pointwise
//          + bias, fused; emits `mid` in channels-last f16  [B][H*W][C].
// Stage 2: weight transform conv_w[oc][c][ky][kx] -> wk[tap][oc][c] (f16).
// Stage 3: implicit GEMM (M=oc=512, N=pixels, K=9*512) with
//          v_wmma_f32_16x16x32_f16; weight tiles (128 oc x 64 c) staged into
//          LDS by the Tensor Data Mover (double buffered, s_wait_tensorcnt),
//          activations read directly from global. Tap-outer loop hoists the
//          reflect-padding address math; 8 WMMAs per barrier interval.
// ---------------------------------------------------------------------------

typedef _Float16    h8   __attribute__((ext_vector_type(8)));
typedef _Float16    v16h __attribute__((ext_vector_type(16)));
typedef float       v8f  __attribute__((ext_vector_type(8)));
typedef unsigned    v4u  __attribute__((ext_vector_type(4)));
typedef int         v8i  __attribute__((ext_vector_type(8)));
typedef int         v4i  __attribute__((ext_vector_type(4)));

#define BB 8
#define CC 512
#define HH 64
#define WW 64
#define HW 4096
#define KSTEPS 72             // 9 taps * (512/64) channel chunks
#define LDSP 72               // halfs per LDS tile row (144B: 128B data + 16B TDM pad)
#define TILE_HALFS (128 * LDSP)
#define TILE_BYTES (TILE_HALFS * 2)

#if defined(__has_builtin)
#if __has_builtin(__builtin_amdgcn_tensor_load_to_lds) && __has_builtin(__builtin_amdgcn_s_wait_tensorcnt)
#define HAS_TDM 1
#endif
#endif
#ifndef HAS_TDM
#define HAS_TDM 0
#endif

// -------------------------- Stage 1: norm + depthwise ----------------------
__global__ __launch_bounds__(256) void norm_dw_kernel(
    const float* __restrict__ x, const float* __restrict__ w_spatial,
    const float* __restrict__ w_pointwise, const float* __restrict__ bias,
    _Float16* __restrict__ mid) {
  const int c = blockIdx.x;
  const int b = blockIdx.y;
  const int t = threadIdx.x;

  __shared__ float raw[HW];
  __shared__ float redA[256];
  __shared__ float redB[256];

  const float* xp = x + ((size_t)(b * CC + c)) * HW;
  float s = 0.f, s2 = 0.f;
#pragma unroll
  for (int i = 0; i < 16; ++i) {
    float v = xp[t + i * 256];
    raw[t + i * 256] = v;
    s += v;
    s2 += v * v;
  }
  redA[t] = s;
  redB[t] = s2;
  __syncthreads();
#pragma unroll
  for (int off = 128; off > 0; off >>= 1) {
    if (t < off) {
      redA[t] += redA[t + off];
      redB[t] += redB[t + off];
    }
    __syncthreads();
  }
  const float mu  = redA[0] * (1.f / HW);
  const float var = redB[0] * (1.f / HW) - mu * mu;
  const float inv = rsqrtf(var + 1e-5f);

  float w[9], Sw = 0.f;
  const float* wsp = w_spatial + ((size_t)(b * CC + c)) * 9;
#pragma unroll
  for (int i = 0; i < 9; ++i) { w[i] = wsp[i]; Sw += w[i]; }
  const float scale = w_pointwise[b * CC + c];
  const float bi    = bias[b * CC + c];

  // depthwise reflect conv on raw, normalization folded in:
  // conv(xn) = inv*(conv(x) - mu*Sw)
#pragma unroll 4
  for (int i = 0; i < 16; ++i) {
    const int p = t + i * 256;
    const int y = p >> 6, xx = p & 63;
    float acc = 0.f;
#pragma unroll
    for (int dy = -1; dy <= 1; ++dy) {
#pragma unroll
      for (int dx = -1; dx <= 1; ++dx) {
        int yy = y + dy; yy = (yy < 0) ? -yy : ((yy > 63) ? 126 - yy : yy);
        int xr = xx + dx; xr = (xr < 0) ? -xr : ((xr > 63) ? 126 - xr : xr);
        acc += w[(dy + 1) * 3 + (dx + 1)] * raw[yy * 64 + xr];
      }
    }
    const float r = ((acc - mu * Sw) * inv) * scale + bi;
    mid[((size_t)b * HW + p) * CC + c] = (_Float16)r;
  }
}

// -------------------------- Stage 2: weight transform ----------------------
__global__ __launch_bounds__(256) void wxform_kernel(
    const float* __restrict__ conv_w, _Float16* __restrict__ wk) {
  const int i = blockIdx.x * 256 + threadIdx.x;   // i = (tap*512 + oc)*512 + c
  if (i >= 9 * CC * CC) return;
  const int c   = i & 511;
  const int oc  = (i >> 9) & 511;
  const int tap = i >> 18;
  wk[i] = (_Float16)conv_w[((size_t)oc * CC + c) * 9 + tap];
}

// -------------------------- TDM staging helpers ----------------------------
#if HAS_TDM
__device__ __forceinline__ void tdm_stage(const _Float16* src, unsigned lds_off) {
  // 2D tile: tile_dim0=64 halfs (contiguous c), tile_dim1=128 rows (oc),
  // row stride 512 halfs, data_size=2B, LDS pad: +4 dwords every 32 dwords
  // (row pitch 128B -> 144B so 16 consecutive rows hit distinct bank groups).
  unsigned long long ga = (unsigned long long)(uintptr_t)src;
  v4u g0 = {1u,                                   // count=1, user mode
            lds_off,                              // lds_addr
            (unsigned)ga,                         // global_addr[31:0]
            (unsigned)(ga >> 32) | (2u << 30)};   // global_addr[56:32] | type=2
  const unsigned long long TD = 1u << 20;         // large dims => no OOB clip
  unsigned long long qa = ((unsigned long long)((1u << 16) |   // data_size=2B
                                                (1u << 20) |   // pad_enable
                                                (4u << 22) |   // pad_interval: 32 dwords
                                                (3u << 25)))   // pad_amount: 4 dwords
                          | ((TD & 0xFFFFull) << 48);          // tensor_dim0[15:0]
  unsigned long long qb = (TD >> 16)                           // tensor_dim0[31:16]
                          | (TD << 16)                         // tensor_dim1
                          | (64ull << 48);                     // tile_dim0 = 64
  unsigned long long qc = 128ull                               // tile_dim1 = 128
                          | (512ull << 32);                    // tensor_dim0_stride
  unsigned long long qd = 0ull;
  v8i g1 = {(int)qa, (int)(qa >> 32), (int)qb, (int)(qb >> 32),
            (int)qc, (int)(qc >> 32), (int)qd, (int)(qd >> 32)};
  v4i z4 = {0, 0, 0, 0};
  v8i z8 = {0, 0, 0, 0, 0, 0, 0, 0};
  __builtin_amdgcn_tensor_load_to_lds(g0, g1, z4, z4, z8, 0);
}
#else
__device__ __forceinline__ void copy_stage(const _Float16* __restrict__ src,
                                           _Float16* __restrict__ dst, int tid) {
  const int r  = tid >> 1;            // 128 rows, 2 threads/row
  const int ho = (tid & 1) * 32;      // 32 halfs each
  const h8* s = (const h8*)(src + (size_t)r * CC + ho);
  h8* d = (h8*)(dst + r * LDSP + ho);
  d[0] = s[0];
  d[1] = s[1];
  d[2] = s[2];
  d[3] = s[3];
}
#endif

// -------------------------- Stage 3: implicit GEMM -------------------------
__global__ __launch_bounds__(256) void gemm_kernel(
    const _Float16* __restrict__ mid, const _Float16* __restrict__ wk,
    const float* __restrict__ conv_b, float* __restrict__ out) {
  const int ocTile = blockIdx.x;          // 0..3  -> 128 ocs
  const int y      = blockIdx.y;          // 0..63 -> one image row (64 pixels)
  const int b      = blockIdx.z;          // 0..7
  const int tid    = threadIdx.x;
  const int lane   = tid & 31;
  const int wid    = tid >> 5;
  const int waveM  = wid >> 1;            // 0..3 : oc quadrant (32 each)
  const int waveN  = wid & 1;             // 0..1 : pixel half (32 each)
  const int l16    = lane >> 4;           // hi/lo half-wave
  const int lr     = lane & 15;

  __shared__ __attribute__((aligned(16))) _Float16 lbuf[2 * TILE_HALFS];
  const unsigned lds_base = (unsigned)(uintptr_t)(&lbuf[0]);
  const int ocBlock = ocTile * 128;

  // per-fragment pixel x positions (N dimension)
  int px[2];
  px[0] = waveN * 32 + lr;
  px[1] = waveN * 32 + 16 + lr;

  v8f acc[2][2];
  const v8f vzero = {0.f, 0.f, 0.f, 0.f, 0.f, 0.f, 0.f, 0.f};
  acc[0][0] = vzero; acc[0][1] = vzero; acc[1][0] = vzero; acc[1][1] = vzero;

  // prologue: stage K-step 0 weight tile
#if HAS_TDM
  if (wid == 0) tdm_stage(wk + (size_t)ocBlock * CC, lds_base);
#else
  copy_stage(wk + (size_t)ocBlock * CC, lbuf, tid);
#endif

  for (int tap = 0; tap < 9; ++tap) {
    // reflect-padded source row/col for this tap: hoisted out of the K loop
    const int dy = tap / 3 - 1, dx = tap % 3 - 1;
    int yy = y + dy; yy = (yy < 0) ? 1 : ((yy > 63) ? 62 : yy);
    const _Float16* bbase[2];
#pragma unroll
    for (int fN = 0; fN < 2; ++fN) {
      int xr = px[fN] + dx; xr = (xr < 0) ? 1 : ((xr > 63) ? 62 : xr);
      bbase[fN] = mid + ((size_t)(b * HW + yy * 64 + xr)) * CC + l16 * 16;
    }

    for (int j = 0; j < 8; ++j) {       // 8 chunks of 64 channels
      const int kk = tap * 8 + j;

      // stage next tile into the other LDS buffer while we compute this one
#if HAS_TDM
      if (wid == 0) {
        if (kk + 1 < KSTEPS) {
          const int nt = (kk + 1) >> 3, nj = (kk + 1) & 7;
          tdm_stage(wk + ((size_t)(nt * CC + ocBlock)) * CC + nj * 64,
                    lds_base + ((kk + 1) & 1) * TILE_BYTES);
          __builtin_amdgcn_s_wait_tensorcnt(1);   // current tile complete
        } else {
          __builtin_amdgcn_s_wait_tensorcnt(0);
        }
      }
#else
      if (kk + 1 < KSTEPS) {
        const int nt = (kk + 1) >> 3, nj = (kk + 1) & 7;
        copy_stage(wk + ((size_t)(nt * CC + ocBlock)) * CC + nj * 64,
                   lbuf + ((kk + 1) & 1) * TILE_HALFS, tid);
      }
#endif
      __syncthreads();

      const _Float16* lt = lbuf + (kk & 1) * TILE_HALFS;

      // two K=32 sub-chunks per staged 64-channel tile -> 8 WMMAs
#pragma unroll
      for (int s = 0; s < 2; ++s) {
        // B operand (activations, direct global): contiguous k order used here (same as A):
        // lane<16 K=0..15, lane>=16 K=16..31, column N = lane%16.
        v16h bf[2];
#pragma unroll
        for (int fN = 0; fN < 2; ++fN) {
          bf[fN] = *(const v16h*)(bbase[fN] + j * 64 + s * 32);
        }
        // A operand (weights from LDS): contiguous k order used here (same as B):
        // lanes<16: K0..15; lanes>=16: K16..31.
#pragma unroll
        for (int fM = 0; fM < 2; ++fM) {
          const int row = waveM * 32 + fM * 16 + lr;
          const _Float16* rp = lt + row * LDSP + s * 32 + l16 * 16;
          union { v16h v; h8 h[2]; } af;
          af.h[0] = *(const h8*)(rp);
          af.h[1] = *(const h8*)(rp + 8);
#pragma unroll
          for (int fN = 0; fN < 2; ++fN) {
            acc[fM][fN] = __builtin_amdgcn_wmma_f32_16x16x32_f16(
                false, af.v, false, bf[fN], (short)0, acc[fM][fN], false, false);
          }
        }
      }
      __syncthreads();   // all reads of lbuf[kk&1] done before it is re-staged
    }
  }

  // epilogue: D 16x16 f32 layout: N = lane%16, M = r + 8*(lane/16)
#pragma unroll
  for (int fM = 0; fM < 2; ++fM) {
#pragma unroll
    for (int fN = 0; fN < 2; ++fN) {
      const int ocb = ocBlock + waveM * 32 + fM * 16 + 8 * l16;
      const int p   = y * 64 + px[fN];
#pragma unroll
      for (int r = 0; r < 8; ++r) {
        const int oc = ocb + r;
        out[((size_t)(b * CC + oc)) * HW + p] = acc[fM][fN][r] + conv_b[oc];
      }
    }
  }
}

// ---------------------------------------------------------------------------
extern "C" void kernel_launch(void* const* d_in, const int* in_sizes, int n_in,
                              void* d_out, int out_size, void* d_ws, size_t ws_size,
                              hipStream_t stream) {
  const float* x      = (const float*)d_in[0];
  const float* w_sp   = (const float*)d_in[1];
  const float* w_pt   = (const float*)d_in[2];
  const float* bias   = (const float*)d_in[3];
  const float* conv_w = (const float*)d_in[4];
  const float* conv_b = (const float*)d_in[5];
  float* out = (float*)d_out;

  _Float16* mid = (_Float16*)d_ws;                              // 32 MB
  _Float16* wk  = (_Float16*)((char*)d_ws + (size_t)BB * HW * CC * 2);  // 4.5 MB

  norm_dw_kernel<<<dim3(CC, BB), 256, 0, stream>>>(x, w_sp, w_pt, bias, mid);

  const int nw = 9 * CC * CC;
  wxform_kernel<<<(nw + 255) / 256, 256, 0, stream>>>(conv_w, wk);

  gemm_kernel<<<dim3(4, HH, BB), 256, 0, stream>>>(mid, wk, conv_b, out);
}
